// RSAL_2680059593313
// MI455X (gfx1250) — matrix operand order repaired and checked by reading, not compiled, not run on hardware
//
#include <hip/hip_runtime.h>
#include <hip/hip_bf16.h>
#include <math.h>

// ---------------------------------------------------------------------------
// SS2D (VMamba) block for MI455X / gfx1250, wave32 + WMMA bf16 + async->LDS
// with double-buffered copy/compute overlap (ASYNCcnt partial waits).
// ---------------------------------------------------------------------------

typedef __bf16 bf16_t;
typedef __attribute__((ext_vector_type(16))) __bf16 v16bf;
typedef __attribute__((ext_vector_type(8)))  float  v8f;

// Exact pointer types for the gfx1250 async->LDS builtins (probe-confirmed).
typedef int v4i_t __attribute__((__vector_size__(16)));
typedef __attribute__((address_space(1))) v4i_t as1_v4i;
typedef __attribute__((address_space(3))) v4i_t as3_v4i;
typedef __attribute__((address_space(1))) int   as1_i32;
typedef __attribute__((address_space(3))) int   as3_i32;

namespace {
constexpr int kDModel = 192;
constexpr int kDState = 16;
constexpr int kDInner = 384;
constexpr int kDtRank = 12;
constexpr int kDirs   = 2;
constexpr int kB = 8, kH = 64, kW = 64;
constexpr int kL   = kH * kW;           // 4096
constexpr int kTok = kB * kL;           // 32768
constexpr int kCReal = kDtRank + 2 * kDState; // 44
constexpr int kCPad  = 48;              // padded x_dbl width
}

// ---- CDNA5 async copy global -> LDS (ASYNCcnt-tracked) ---------------------
__device__ __forceinline__ void async_ld_b128(void* lds, const void* g) {
  __builtin_amdgcn_global_load_async_to_lds_b128(
      (as1_v4i*)(uintptr_t)g, (as3_v4i*)(uint32_t)(uintptr_t)lds, 0, 0);
}

__device__ __forceinline__ void async_ld_b32(void* lds, const void* g) {
  __builtin_amdgcn_global_load_async_to_lds_b32(
      (as1_i32*)(uintptr_t)g, (as3_i32*)(uint32_t)(uintptr_t)lds, 0, 0);
}

__device__ __forceinline__ void wait_async0() {
  __builtin_amdgcn_s_wait_asynccnt(0);
}
__device__ __forceinline__ void wait_async1() {
  __builtin_amdgcn_s_wait_asynccnt(1); // oldest copy done; newest may still fly
}

// ---- WMMA tile helpers (16x16x32 bf16, wave32 layouts per CDNA5 ISA 7.12.2)
// A fragment from an LDS-staged 16x32 fp32 tile (row-major [16][32]).
// element j -> K = (j&7) + ((j>>3)<<4) + ((lane>>4)<<3)
__device__ __forceinline__ v16bf frag_a_from_lds(const float* __restrict__ sA,
                                                 int lane) {
  const float* r = sA + ((lane & 15) << 5) + ((lane >> 4) << 3);
  v16bf a;
#pragma unroll
  for (int j = 0; j < 8; ++j) a[j] = (bf16_t)r[j];
#pragma unroll
  for (int j = 0; j < 8; ++j) a[8 + j] = (bf16_t)r[16 + j];
  return a;
}

// B (32x16, KxN) from row-major bf16 weight W[n][ldk]: lane n = lane&15;
// element j -> K = kbase + (j&7) + ((j>>3)<<4) + ((lane>>4)<<3)
__device__ __forceinline__ v16bf load_b_tile_bf16(const bf16_t* __restrict__ W,
                                                  int ldk, int tileN, int kbase,
                                                  int lane) {
  const bf16_t* row = W + (size_t)(tileN * 16 + (lane & 15)) * ldk;
  const int kb = kbase + ((lane >> 4) << 3);
  v16bf b;
#pragma unroll
  for (int j = 0; j < 8; ++j) b[j] = row[kb + j];
#pragma unroll
  for (int j = 0; j < 8; ++j) b[8 + j] = row[kb + 16 + j];
  return b;
}

// Cooperative async stage of one 16x32 fp32 A tile into LDS (2 KB, 128 x b128).
template <int NTHREADS>
__device__ __forceinline__ void stage_a_async(float* __restrict__ sA,
                                              const float* __restrict__ A,
                                              int ldk, int tileM, int kbase,
                                              int tid) {
  for (int i = tid; i < 128; i += NTHREADS) {
    const int row = i >> 3;
    const int col = (i & 7) << 2;
    const float* g = A + (size_t)(tileM * 16 + row) * ldk + kbase + col;
    async_ld_b128(&sA[i << 2], g);
  }
}

// Double-buffered K-loop shared by the three GEMM kernels.
template <int NTHREADS, int NK>
__device__ __forceinline__ v8f gemm_k_loop(const float* __restrict__ A, int ldk,
                                           const bf16_t* __restrict__ W,
                                           float (*sA)[512], int tileM, int tileN,
                                           int tid, int lane) {
  v8f c = {0.f, 0.f, 0.f, 0.f, 0.f, 0.f, 0.f, 0.f};
  stage_a_async<NTHREADS>(sA[0], A, ldk, tileM, 0, tid);
#pragma unroll 1
  for (int s = 0; s < NK; ++s) {
    if (s + 1 < NK) {
      stage_a_async<NTHREADS>(sA[(s + 1) & 1], A, ldk, tileM, (s + 1) * 32, tid);
      wait_async1();
    } else {
      wait_async0();
    }
    __syncthreads();
    v16bf a = frag_a_from_lds(sA[s & 1], lane);
    v16bf b = load_b_tile_bf16(W, ldk, tileN, s * 32, lane);
    c = __builtin_amdgcn_wmma_f32_16x16x32_bf16(false, a, false, b, (short)0, c,
                                                false, false);
    __syncthreads(); // readers done before this buffer is restaged next step
  }
  return c;
}

// ---------------------------------------------------------------------------
// K0: weight conversion fp32->bf16 (+ x_proj zero-pad 44->48) and A = -exp(Alog)
__global__ void k_prep(const float* __restrict__ wi, const float* __restrict__ wx,
                       const float* __restrict__ wo, const float* __restrict__ alogs,
                       bf16_t* __restrict__ wi_b, bf16_t* __restrict__ wx_b,
                       bf16_t* __restrict__ wo_b, float* __restrict__ As) {
  const int i = blockIdx.x * blockDim.x + threadIdx.x;
  if (i < 2 * kDInner * kDModel) wi_b[i] = (bf16_t)wi[i];
  if (i < kDirs * kCPad * kDInner) {
    const int d = i % kDInner;
    const int c = (i / kDInner) % kCPad;
    const int k = i / (kDInner * kCPad);
    float v = (c < kCReal) ? wx[((size_t)k * kCReal + c) * kDInner + d] : 0.0f;
    wx_b[i] = (bf16_t)v;
  }
  if (i < kDModel * kDInner) wo_b[i] = (bf16_t)wo[i];
  if (i < kDirs * kDInner * kDState) As[i] = -__expf(alogs[i]);
}

// ---------------------------------------------------------------------------
// K1: in_proj GEMM [32768x192]@[192x768]. 8 waves share tileM (one tileN each).
__global__ void __launch_bounds__(256) k_in_proj(
    const float* __restrict__ x, const bf16_t* __restrict__ wi,
    float* __restrict__ xx, float* __restrict__ zz) {
  const int tid  = threadIdx.x;
  const int lane = tid & 31;
  const int tileM = blockIdx.x / 6;
  const int tileN = (blockIdx.x % 6) * 8 + (tid >> 5);
  __shared__ float sA[2][512];
  v8f c = gemm_k_loop<256, kDModel / 32>(x, kDModel, wi, sA, tileM, tileN, tid, lane);
  const int n  = tileN * 16 + (lane & 15);
  const int m0 = tileM * 16 + ((lane >> 4) << 3);
#pragma unroll
  for (int r = 0; r < 8; ++r) {
    const size_t tok = (size_t)(m0 + r);
    if (n < kDInner) xx[tok * kDInner + n]             = c[r];
    else             zz[tok * kDInner + (n - kDInner)] = c[r];
  }
}

// ---------------------------------------------------------------------------
// K2: depthwise 3x3 conv (zero pad) + bias + SiLU, NHWC
__global__ void k_conv(const float* __restrict__ xx, const float* __restrict__ cw,
                       const float* __restrict__ cb, float* __restrict__ xc) {
  const int idx = blockIdx.x * blockDim.x + threadIdx.x;
  if (idx >= kTok * kDInner) return;
  const int c   = idx % kDInner;
  const int tok = idx / kDInner;
  const int w = tok & 63, h = (tok >> 6) & 63, b = tok >> 12;
  float acc = cb[c];
#pragma unroll
  for (int i = 0; i < 3; ++i) {
    const int hh = h + i - 1;
    if ((unsigned)hh >= (unsigned)kH) continue;
#pragma unroll
    for (int j = 0; j < 3; ++j) {
      const int ww = w + j - 1;
      if ((unsigned)ww >= (unsigned)kW) continue;
      const size_t src = (size_t)((b << 12) + (hh << 6) + ww) * kDInner + c;
      acc = fmaf(xx[src], cw[c * 9 + i * 3 + j], acc);
    }
  }
  xc[(size_t)tok * kDInner + c] = acc / (1.0f + __expf(-acc)); // SiLU
}

// ---------------------------------------------------------------------------
// K3: x_proj GEMM per direction. 3 waves share tileM (one tileN each).
__global__ void __launch_bounds__(96) k_x_proj(
    const float* __restrict__ xc, const bf16_t* __restrict__ wx,
    float* __restrict__ xdbl) {
  const int k    = blockIdx.y;
  const int tid  = threadIdx.x;
  const int lane = tid & 31;
  const int tileM = blockIdx.x;
  const int tileN = tid >> 5;
  const bf16_t* w = wx + (size_t)k * kCPad * kDInner;
  __shared__ float sA[2][512];
  v8f c = gemm_k_loop<96, kDInner / 32>(xc, kDInner, w, sA, tileM, tileN, tid, lane);
  const int n  = tileN * 16 + (lane & 15);
  const int m0 = tileM * 16 + ((lane >> 4) << 3);
  float* outk = xdbl + (size_t)k * kTok * kCPad;
#pragma unroll
  for (int r = 0; r < 8; ++r) outk[(size_t)(m0 + r) * kCPad + n] = c[r];
}

// ---------------------------------------------------------------------------
// K4: fused dt-projection + softplus + selective scan, double-buffered row
// broadcast (async->LDS) + register double buffer for u.
__global__ void __launch_bounds__(128) k_scan(
    const float* __restrict__ xc, const float* __restrict__ xdbl,
    const float* __restrict__ dtw, const float* __restrict__ dtb,
    const float* __restrict__ As, const float* __restrict__ Ds,
    float* __restrict__ ym) {
  const int gid    = blockIdx.x;
  const int dchunk = gid % 3;
  const int k      = (gid / 3) % kDirs;
  const int b      = gid / (3 * kDirs);
  const int d      = dchunk * 128 + threadIdx.x;
  const int kd     = k * kDInner + d;

  float Arow[kDState];
#pragma unroll
  for (int n = 0; n < kDState; ++n) Arow[n] = As[(size_t)kd * kDState + n];
  float wdt[kDtRank];
#pragma unroll
  for (int r = 0; r < kDtRank; ++r) wdt[r] = dtw[(size_t)kd * kDtRank + r];
  const float bias = dtb[kd];
  const float Dv   = Ds[kd];

  float h[kDState];
#pragma unroll
  for (int n = 0; n < kDState; ++n) h[n] = 0.f;

  __shared__ float s[2][kCPad];
  const float* xd  = xdbl + ((size_t)k * kTok + (size_t)b * kL) * kCPad;
  const float* xcb = xc + (size_t)b * kL * kDInner;
  float* ymb       = ym + (size_t)b * kL * kDInner;

  const int l0 = k ? (kL - 1) : 0;
  if (threadIdx.x < kCReal)
    async_ld_b32(&s[0][threadIdx.x], &xd[(size_t)l0 * kCPad + threadIdx.x]);
  float u_cur = xcb[(size_t)l0 * kDInner + d];

  for (int t = 0; t < kL; ++t) {
    const int l = k ? (kL - 1 - t) : t;
    float u_nxt = 0.f;
    if (t + 1 < kL) {
      const int ln = k ? (kL - 2 - t) : (t + 1);
      if (threadIdx.x < kCReal)
        async_ld_b32(&s[(t + 1) & 1][threadIdx.x],
                     &xd[(size_t)ln * kCPad + threadIdx.x]);
      u_nxt = xcb[(size_t)ln * kDInner + d];
      wait_async1();
    } else {
      wait_async0();
    }
    __syncthreads();

    const float* sr = s[t & 1];
    float draw = bias;
#pragma unroll
    for (int r = 0; r < kDtRank; ++r) draw = fmaf(sr[r], wdt[r], draw);
    const float dl  = (draw > 20.f) ? draw : log1pf(__expf(draw)); // softplus
    const float dlu = dl * u_cur;

    float y = 0.f;
#pragma unroll
    for (int n = 0; n < kDState; ++n) {
      const float dA = __expf(dl * Arow[n]);
      h[n] = fmaf(dA, h[n], dlu * sr[kDtRank + n]);
      y    = fmaf(h[n], sr[kDtRank + kDState + n], y);
    }
    atomicAdd(&ymb[(size_t)l * kDInner + d], fmaf(Dv, u_cur, y));
    __syncthreads(); // readers done before this buffer is restaged next step
    u_cur = u_nxt;

    // prefetch two steps ahead (lowers to global_prefetch_b8)
    const int lp = k ? (kL - 3 - t) : (t + 2);
    if (threadIdx.x == 0 && (unsigned)lp < (unsigned)kL)
      __builtin_prefetch(&xd[(size_t)lp * kCPad], 0, 1);
  }
}

// ---------------------------------------------------------------------------
// K5: LayerNorm over d=384 + erf-GELU gate; gates z in place -> yg
__global__ void __launch_bounds__(128) k_ln_gate(
    const float* __restrict__ ym, float* __restrict__ zz,
    const float* __restrict__ lnw, const float* __restrict__ lnb) {
  const int tok = blockIdx.x;
  const float* y = ym + (size_t)tok * kDInner;
  float* zrow    = zz + (size_t)tok * kDInner;

  const float v0 = y[threadIdx.x];
  const float v1 = y[threadIdx.x + 128];
  const float v2 = y[threadIdx.x + 256];
  float s  = v0 + v1 + v2;
  float ss = v0 * v0 + v1 * v1 + v2 * v2;
#pragma unroll
  for (int off = 16; off; off >>= 1) {
    s  += __shfl_xor(s, off, 32);
    ss += __shfl_xor(ss, off, 32);
  }
  __shared__ float ps[4], pss[4];
  __shared__ float mu_s, rs_s;
  const int wid = threadIdx.x >> 5;
  if ((threadIdx.x & 31) == 0) { ps[wid] = s; pss[wid] = ss; }
  __syncthreads();
  if (threadIdx.x == 0) {
    const float S  = ps[0] + ps[1] + ps[2] + ps[3];
    const float SS = pss[0] + pss[1] + pss[2] + pss[3];
    const float mu = S / (float)kDInner;
    const float var = SS / (float)kDInner - mu * mu;
    mu_s = mu;
    rs_s = rsqrtf(var + 1e-5f);
  }
  __syncthreads();
  const float mu = mu_s, rs = rs_s;
#pragma unroll
  for (int p = 0; p < 3; ++p) {
    const int d = threadIdx.x + p * 128;
    const float yv = (p == 0) ? v0 : ((p == 1) ? v1 : v2);
    const float yn = (yv - mu) * rs * lnw[d] + lnb[d];
    const float zv = zrow[d];
    const float g  = 0.5f * zv * (1.0f + erff(zv * 0.70710678118654752f));
    zrow[d] = yn * g;
  }
}

// ---------------------------------------------------------------------------
// K6: out_proj GEMM [32768x384]@[384x192]. 12 waves share tileM (all tileN).
__global__ void __launch_bounds__(384) k_out_proj(
    const float* __restrict__ yg, const bf16_t* __restrict__ wo,
    float* __restrict__ out) {
  const int tid  = threadIdx.x;
  const int lane = tid & 31;
  const int tileM = blockIdx.x;
  const int tileN = tid >> 5;
  __shared__ float sA[2][512];
  v8f c = gemm_k_loop<384, kDInner / 32>(yg, kDInner, wo, sA, tileM, tileN, tid, lane);
  const int n  = tileN * 16 + (lane & 15);
  const int m0 = tileM * 16 + ((lane >> 4) << 3);
#pragma unroll
  for (int r = 0; r < 8; ++r) out[(size_t)(m0 + r) * kDModel + n] = c[r];
}

// ---------------------------------------------------------------------------
extern "C" void kernel_launch(void* const* d_in, const int* in_sizes, int n_in,
                              void* d_out, int out_size, void* d_ws, size_t ws_size,
                              hipStream_t stream) {
  const float* x     = (const float*)d_in[0];   // [8,64,64,192]
  const float* wi    = (const float*)d_in[1];   // [768,192]
  const float* cw    = (const float*)d_in[2];   // [384,1,3,3]
  const float* cb    = (const float*)d_in[3];   // [384]
  const float* wx    = (const float*)d_in[4];   // [2,44,384]
  const float* dtw   = (const float*)d_in[5];   // [2,384,12]
  const float* dtb   = (const float*)d_in[6];   // [2,384]
  const float* alogs = (const float*)d_in[7];   // [768,16]
  const float* Ds    = (const float*)d_in[8];   // [768]
  const float* lnw   = (const float*)d_in[9];   // [384]
  const float* lnb   = (const float*)d_in[10];  // [384]
  const float* wo    = (const float*)d_in[11];  // [192,384]
  float* out = (float*)d_out;

  char* ws = (char*)d_ws;
  size_t off = 0;
  auto carve = [&](size_t bytes) -> char* {
    char* p = ws + off;
    off += (bytes + 255) & ~(size_t)255;
    return p;
  };
  bf16_t* wi_b = (bf16_t*)carve((size_t)2 * kDInner * kDModel * sizeof(bf16_t));
  bf16_t* wx_b = (bf16_t*)carve((size_t)kDirs * kCPad * kDInner * sizeof(bf16_t));
  bf16_t* wo_b = (bf16_t*)carve((size_t)kDModel * kDInner * sizeof(bf16_t));
  float*  As   = (float*) carve((size_t)kDirs * kDInner * kDState * sizeof(float));
  float*  xx   = (float*) carve((size_t)kTok * kDInner * sizeof(float)); // reused as ym
  float*  zz   = (float*) carve((size_t)kTok * kDInner * sizeof(float)); // gated in place
  float*  xc   = (float*) carve((size_t)kTok * kDInner * sizeof(float));
  float*  xdbl = (float*) carve((size_t)kDirs * kTok * kCPad * sizeof(float));
  if (off > ws_size) return; // workspace too small: bail rather than corrupt

  // K0: weight prep
  {
    const int n = 2 * kDInner * kDModel; // largest converted array
    k_prep<<<dim3((n + 255) / 256), dim3(256), 0, stream>>>(wi, wx, wo, alogs,
                                                            wi_b, wx_b, wo_b, As);
  }
  // K1: in_proj GEMM (2048 tileM x 6 N-chunks, 8 waves/block)
  k_in_proj<<<dim3((kTok / 16) * 6), dim3(256), 0, stream>>>(x, wi_b, xx, zz);
  // K2: depthwise conv + SiLU
  k_conv<<<dim3((kTok * kDInner) / 256), dim3(256), 0, stream>>>(xx, cw, cb, xc);
  // K3: x_proj GEMM per direction (2048 tileM, 3 waves/block)
  k_x_proj<<<dim3(kTok / 16, kDirs), dim3(96), 0, stream>>>(xc, wx_b, xdbl);
  // zero the direction-merge accumulator (xx is free after the conv)
  (void)hipMemsetAsync(xx, 0, (size_t)kTok * kDInner * sizeof(float), stream);
  // K4: fused dt-proj + softplus + selective scan (both directions, atomic merge)
  k_scan<<<dim3(kB * kDirs * 3), dim3(128), 0, stream>>>(xc, xdbl, dtw, dtb, As, Ds, xx);
  // K5: LayerNorm + GELU gate (z gated in place -> yg)
  k_ln_gate<<<dim3(kTok), dim3(128), 0, stream>>>(xx, zz, lnw, lnb);
  // K6: out_proj GEMM (2048 tileM, 12 waves/block)
  k_out_proj<<<dim3(kTok / 16), dim3(384), 0, stream>>>(zz, wo_b, out);
}
